// SuperResolutionBlock_56719338111572
// MI455X (gfx1250) — matrix operand order repaired and checked by reading, not compiled, not run on hardware
//
#include <hip/hip_runtime.h>
#include <math.h>

typedef __attribute__((ext_vector_type(16))) _Float16 v16h;
typedef __attribute__((ext_vector_type(8)))  _Float16 v8h;
typedef __attribute__((ext_vector_type(8)))  float    v8f;

#define WMMA_F32_F16(a, b, c) \
  __builtin_amdgcn_wmma_f32_16x16x32_f16(false, (a), false, (b), (short)0, (c), false, false)

__device__ __forceinline__ int imax(int a, int b) { return a > b ? a : b; }
__device__ __forceinline__ int imin(int a, int b) { return a < b ? a : b; }

// A-fragment (16x32 f16, contiguous k = 16h + s) from LDS row-major [32][kstride] halves.
// lane<16: halves 0..15 -> K=kbase+0..15
// lane>=16: shifted by +16.  Two aligned 16B ds loads.
__device__ __forceinline__ v16h load_a_frag(const _Float16* sW, int row, int kbase,
                                            int kstride, int lane) {
  const int offA = (lane >> 4) << 4;  // 0 or 16
  const _Float16* p = sW + row * kstride + kbase + offA;
  v8h lo = *(const v8h*)(p);
  v8h hi = *(const v8h*)(p + 8);
  v16h a;
#pragma unroll
  for (int i = 0; i < 8; ++i) { a[i] = lo[i]; a[i + 8] = hi[i]; }
  return a;
}

// ---------------------------------------------------------------------------
// Implicit-GEMM 3x3 conv (pad=1). M out channels (<=32), C in channels (even).
// K padded to 16 per channel: chunk of 32 = 2 channels, one per lane-half.
// Tap indices are clamped + masked (branch-free): all loads unconditional.
// ---------------------------------------------------------------------------
__global__ __launch_bounds__(256) void k_conv3x3_wmma(
    const float* __restrict__ in, const float* __restrict__ w,
    const float* __restrict__ bias, float* __restrict__ out,
    int C, int M, int H, int W, int tiles) {
  __shared__ __attribute__((aligned(16))) _Float16 sW[32 * 64 * 16];  // 64 KB max
  const int C16 = C * 16;
  for (int i = threadIdx.x; i < 32 * C16; i += blockDim.x) {
    int m = i / C16, k = i - m * C16;
    int c = k >> 4, kk = k & 15;
    float v = 0.f;
    if (m < M && kk < 9) v = w[(m * C + c) * 9 + kk];
    sW[i] = (_Float16)v;
  }
  __syncthreads();

  const int lane = threadIdx.x & 31;
  const int n = lane & 15;
  const int HW = H * W;
  const int nw = (gridDim.x * blockDim.x) >> 5;
  for (int t = (blockIdx.x * blockDim.x + threadIdx.x) >> 5; t < tiles; t += nw) {
    const int flat = t << 4;
    const int x0 = flat % W;
    const int y = (flat / W) % H;
    const int b = flat / HW;
    const int xp = x0 + n;

    // Tap indices/masks invariant across channels: hoist out of the K loop.
    int cidx[9];
    float msk[9];
#pragma unroll
    for (int kk = 0; kk < 9; ++kk) {
      const int yy = y + kk / 3 - 1;
      const int xx = xp + kk % 3 - 1;
      const bool ok = (yy >= 0) && (yy < H) && (xx >= 0) && (xx < W);
      cidx[kk] = imin(imax(yy, 0), H - 1) * W + imin(imax(xx, 0), W - 1);
      msk[kk] = ok ? 1.f : 0.f;
    }

    // Each lane owns one channel per chunk: lanes 16..31 take c0+1.
    const float* pbase = in + (size_t)(b * C + (lane >> 4)) * HW;

    v8f acc0 = {}, acc1 = {};
    for (int c0 = 0; c0 < C; c0 += 2) {
      if (c0 + 2 < C) __builtin_prefetch(pbase + 2 * HW + cidx[4], 0, 1);
      // Issue all 9 tap loads unconditionally, then combine.
      float vals[9];
#pragma unroll
      for (int kk = 0; kk < 9; ++kk) vals[kk] = pbase[cidx[kk]];
      v16h bf = {};
#pragma unroll
      for (int kk = 0; kk < 9; ++kk) bf[kk] = (_Float16)(vals[kk] * msk[kk]);

      const int kbase = c0 << 4;
      v16h a0 = load_a_frag(sW, n, kbase, C16, lane);
      v16h a1 = load_a_frag(sW, n + 16, kbase, C16, lane);
      acc0 = WMMA_F32_F16(a0, bf, acc0);
      acc1 = WMMA_F32_F16(a1, bf, acc1);
      pbase += 2 * HW;
    }

    const int mb = (lane >> 4) << 3;  // D layout: lanes>=16 hold rows M=8..15
#pragma unroll
    for (int vi = 0; vi < 8; ++vi) {
      const int m0 = mb + vi;
      if (m0 < M)
        out[((size_t)(b * M + m0) * H + y) * W + xp] = acc0[vi] + bias[m0];
      const int m1 = 16 + mb + vi;
      if (m1 < M)
        out[((size_t)(b * M + m1) * H + y) * W + xp] = acc1[vi] + bias[m1];
    }
  }
}

// ---------------------------------------------------------------------------
// Deformable conv 3x3: per-pixel bilinear gather (mask+valid folded into tap
// weights), then WMMA einsum over K = C*9 (padded to C*16). 32 out channels.
// ELU fused on output; addOut fuses the residual accumulation.
// ---------------------------------------------------------------------------
__global__ __launch_bounds__(256) void k_dcn_wmma(
    const float* __restrict__ in, const float* __restrict__ om,
    const float* __restrict__ w, const float* __restrict__ bias,
    float* __restrict__ out, int C, int H, int W, int tiles, int addOut) {
  __shared__ __attribute__((aligned(16))) _Float16 sW[32 * 64 * 16];
  const int C16 = C * 16;
  for (int i = threadIdx.x; i < 32 * C16; i += blockDim.x) {
    int m = i / C16, k = i - m * C16;
    int c = k >> 4, kk = k & 15;
    float v = 0.f;
    if (kk < 9) v = w[(m * C + c) * 9 + kk];
    sW[i] = (_Float16)v;
  }
  __syncthreads();

  const int lane = threadIdx.x & 31;
  const int n = lane & 15;
  const int HW = H * W;
  const int nw = (gridDim.x * blockDim.x) >> 5;
  for (int t = (blockIdx.x * blockDim.x + threadIdx.x) >> 5; t < tiles; t += nw) {
    const int flat = t << 4;
    const int x0 = flat % W;
    const int y = (flat / W) % H;
    const int b = flat / HW;
    const int xp = x0 + n;

    // Decode offsets + mask for this pixel: 4 clamped tap indices + weights/kk.
    const float* omp = om + (size_t)b * 27 * HW + (size_t)y * W + xp;
    int i00[9], i01[9], i10[9], i11[9];
    float w00[9], w01[9], w10[9], w11[9];
#pragma unroll
    for (int kk = 0; kk < 9; ++kk) {
      const float dyv = omp[(size_t)(2 * kk) * HW];
      const float dxv = omp[(size_t)(2 * kk + 1) * HW];
      float mk = omp[(size_t)(18 + kk) * HW];
      mk = 1.f / (1.f + __expf(-mk));  // sigmoid mask
      const float py = (float)y + (float)(kk / 3 - 1) + dyv;
      const float px = (float)xp + (float)(kk % 3 - 1) + dxv;
      const float fy = floorf(py), fx = floorf(px);
      const float ty = py - fy, tx = px - fx;
      const int iy0 = (int)fy, ix0 = (int)fx;
      const int iy1 = iy0 + 1, ix1 = ix0 + 1;
      const float vy0 = (iy0 >= 0 && iy0 < H) ? 1.f : 0.f;
      const float vy1 = (iy1 >= 0 && iy1 < H) ? 1.f : 0.f;
      const float vx0 = (ix0 >= 0 && ix0 < W) ? 1.f : 0.f;
      const float vx1 = (ix1 >= 0 && ix1 < W) ? 1.f : 0.f;
      const int cy0 = imin(imax(iy0, 0), H - 1), cy1 = imin(imax(iy1, 0), H - 1);
      const int cx0 = imin(imax(ix0, 0), W - 1), cx1 = imin(imax(ix1, 0), W - 1);
      i00[kk] = cy0 * W + cx0; i01[kk] = cy0 * W + cx1;
      i10[kk] = cy1 * W + cx0; i11[kk] = cy1 * W + cx1;
      w00[kk] = (1.f - ty) * (1.f - tx) * mk * vy0 * vx0;
      w01[kk] = (1.f - ty) * tx * mk * vy0 * vx1;
      w10[kk] = ty * (1.f - tx) * mk * vy1 * vx0;
      w11[kk] = ty * tx * mk * vy1 * vx1;
    }

    const float* pbase = in + (size_t)(b * C + (lane >> 4)) * HW;

    v8f acc0 = {}, acc1 = {};
    for (int c0 = 0; c0 < C; c0 += 2) {
      if (c0 + 2 < C) __builtin_prefetch(pbase + 2 * HW + i00[4], 0, 1);
      // Issue all 36 gathers unconditionally first, then combine.
      float t00[9], t01[9], t10[9], t11[9];
#pragma unroll
      for (int kk = 0; kk < 9; ++kk) {
        t00[kk] = pbase[i00[kk]];
        t01[kk] = pbase[i01[kk]];
        t10[kk] = pbase[i10[kk]];
        t11[kk] = pbase[i11[kk]];
      }
      v16h bf = {};
#pragma unroll
      for (int kk = 0; kk < 9; ++kk) {
        const float v = w00[kk] * t00[kk] + w01[kk] * t01[kk] +
                        w10[kk] * t10[kk] + w11[kk] * t11[kk];
        bf[kk] = (_Float16)v;
      }
      const int kbase = c0 << 4;
      v16h a0 = load_a_frag(sW, n, kbase, C16, lane);
      v16h a1 = load_a_frag(sW, n + 16, kbase, C16, lane);
      acc0 = WMMA_F32_F16(a0, bf, acc0);
      acc1 = WMMA_F32_F16(a1, bf, acc1);
      pbase += 2 * HW;
    }

    const int mb = (lane >> 4) << 3;
#pragma unroll
    for (int vi = 0; vi < 8; ++vi) {
      const int m0 = mb + vi;
      const size_t o0 = ((size_t)(b * 32 + m0) * H + y) * W + xp;
      float v0 = acc0[vi] + bias[m0];
      v0 = v0 > 0.f ? v0 : (__expf(v0) - 1.f);  // ELU
      if (addOut) v0 += out[o0];
      out[o0] = v0;
      const int m1 = 16 + mb + vi;
      const size_t o1 = ((size_t)(b * 32 + m1) * H + y) * W + xp;
      float v1 = acc1[vi] + bias[m1];
      v1 = v1 > 0.f ? v1 : (__expf(v1) - 1.f);
      if (addOut) v1 += out[o1];
      out[o1] = v1;
    }
  }
}

// ---------------------------------------------------------------------------
// Residual 1x1 conv: 64 -> 32 @128x128. K = 64 = two WMMA chunks.
// ---------------------------------------------------------------------------
__global__ __launch_bounds__(256) void k_conv1x1_wmma(
    const float* __restrict__ in, const float* __restrict__ w,
    const float* __restrict__ bias, float* __restrict__ out, int tiles) {
  const int C = 64, H = 128, W = 128, HW = H * W;
  __shared__ __attribute__((aligned(16))) _Float16 sW[32 * 64];
  for (int i = threadIdx.x; i < 32 * 64; i += blockDim.x) sW[i] = (_Float16)w[i];
  __syncthreads();

  const int lane = threadIdx.x & 31;
  const int n = lane & 15;
  const int nw = (gridDim.x * blockDim.x) >> 5;
  for (int t = (blockIdx.x * blockDim.x + threadIdx.x) >> 5; t < tiles; t += nw) {
    const int flat = t << 4;
    const int x0 = flat % W;
    const int y = (flat / W) % H;
    const int b = flat / HW;
    const int xp = x0 + n;

    v8f acc0 = {}, acc1 = {};
#pragma unroll
    for (int kb = 0; kb < 64; kb += 32) {
      const int cb = kb + ((lane >> 4) << 4);  // lanes>=16: K = kb+16..31
      const float* p = in + (size_t)(b * C + cb) * HW + y * W + xp;
      float vals[16];
#pragma unroll
      for (int h = 0; h < 16; ++h) vals[h] = p[(size_t)h * HW];
      v16h bf;
#pragma unroll
      for (int h = 0; h < 16; ++h) bf[h] = (_Float16)vals[h];
      v16h a0 = load_a_frag(sW, n, kb, 64, lane);
      v16h a1 = load_a_frag(sW, n + 16, kb, 64, lane);
      acc0 = WMMA_F32_F16(a0, bf, acc0);
      acc1 = WMMA_F32_F16(a1, bf, acc1);
    }

    const int mb = (lane >> 4) << 3;
#pragma unroll
    for (int vi = 0; vi < 8; ++vi) {
      const int m0 = mb + vi;
      out[((size_t)(b * 32 + m0) * H + y) * W + xp] = acc0[vi] + bias[m0];
      const int m1 = 16 + mb + vi;
      out[((size_t)(b * 32 + m1) * H + y) * W + xp] = acc1[vi] + bias[m1];
    }
  }
}

// ---------------------------------------------------------------------------
// 2x bilinear upsample (half-pixel centers, edge clamp) — jax.image.resize.
// ---------------------------------------------------------------------------
__global__ __launch_bounds__(256) void k_up2(const float* __restrict__ src,
                                             float* __restrict__ dst,
                                             int BC, int Hin, int Win) {
  const int Hout = Hin * 2, Wout = Win * 2;
  const int total = BC * Hout * Wout;
  for (int i = blockIdx.x * blockDim.x + threadIdx.x; i < total;
       i += gridDim.x * blockDim.x) {
    const int ox = i % Wout;
    const int oy = (i / Wout) % Hout;
    const int pc = i / (Wout * Hout);
    const float cy = (oy + 0.5f) * 0.5f - 0.5f;
    const float cx = (ox + 0.5f) * 0.5f - 0.5f;
    int y0 = (int)floorf(cy); const float ty = cy - (float)y0;
    int x0 = (int)floorf(cx); const float tx = cx - (float)x0;
    const int y1 = imin(y0 + 1, Hin - 1); y0 = imax(y0, 0);
    const int x1 = imin(x0 + 1, Win - 1); x0 = imax(x0, 0);
    const float* p = src + (size_t)pc * Hin * Win;
    const float v00 = p[y0 * Win + x0], v01 = p[y0 * Win + x1];
    const float v10 = p[y1 * Win + x0], v11 = p[y1 * Win + x1];
    dst[i] = (1.f - ty) * ((1.f - tx) * v00 + tx * v01) +
             ty * ((1.f - tx) * v10 + tx * v11);
  }
}

// ---------------------------------------------------------------------------
extern "C" void kernel_launch(void* const* d_in, const int* in_sizes, int n_in,
                              void* d_out, int out_size, void* d_ws, size_t ws_size,
                              hipStream_t stream) {
  (void)in_sizes; (void)n_in; (void)out_size; (void)ws_size;
  const float* x   = (const float*)d_in[0];
  const float* w1  = (const float*)d_in[1];
  const float* b1  = (const float*)d_in[2];
  const float* ow1 = (const float*)d_in[3];
  const float* ob1 = (const float*)d_in[4];
  const float* w2  = (const float*)d_in[5];
  const float* b2  = (const float*)d_in[6];
  const float* ow2 = (const float*)d_in[7];
  const float* ob2 = (const float*)d_in[8];
  const float* rw  = (const float*)d_in[9];
  const float* rb  = (const float*)d_in[10];
  float* out = (float*)d_out;

  // Workspace layout (floats)
  float* ws  = (float*)d_ws;
  float* om1 = ws;                 // 2*27*128*128 = 884736
  float* h1  = om1 + 884736;       // 2*32*128*128 = 1048576
  float* h1u = h1 + 1048576;       // 2*32*256*256 = 4194304
  float* om2 = h1u + 4194304;      // 2*27*256*256 = 3538944
  float* r   = om2 + 3538944;      // 2*32*128*128 = 1048576

  const int tiles1 = 2 * 128 * 128 / 16;  // 2048
  const int tiles2 = 2 * 256 * 256 / 16;  // 8192

  // 1. om1 = conv3x3(x; ow1, ob1)           (64 -> 27, 128x128)
  k_conv3x3_wmma<<<tiles1 / 8, 256, 0, stream>>>(x, ow1, ob1, om1, 64, 27, 128, 128, tiles1);
  // 2. h1 = ELU(dcn(x, om1; w1, b1))        (64 -> 32, 128x128)
  k_dcn_wmma<<<tiles1 / 8, 256, 0, stream>>>(x, om1, w1, b1, h1, 64, 128, 128, tiles1, 0);
  // 3. h1u = up2(h1)                        (256x256)
  k_up2<<<4096, 256, 0, stream>>>(h1, h1u, 2 * 32, 128, 128);
  // 4. om2 = conv3x3(h1u; ow2, ob2)         (32 -> 27, 256x256)
  k_conv3x3_wmma<<<tiles2 / 8, 256, 0, stream>>>(h1u, ow2, ob2, om2, 32, 27, 256, 256, tiles2);
  // 5. r = conv1x1(x; rw, rb)               (64 -> 32, 128x128)
  k_conv1x1_wmma<<<tiles1 / 8, 256, 0, stream>>>(x, rw, rb, r, tiles1);
  // 6. out = up2(r)
  k_up2<<<4096, 256, 0, stream>>>(r, out, 2 * 32, 128, 128);
  // 7. out += ELU(dcn(h1u, om2; w2, b2))    (32 -> 32, 256x256)
  k_dcn_wmma<<<tiles2 / 8, 256, 0, stream>>>(h1u, om2, w2, b2, out, 32, 256, 256, tiles2, 1);
}
